// VQ_26087631356273
// MI455X (gfx1250) — matrix operand order repaired and checked by reading, not compiled, not run on hardware
//
#include <hip/hip_runtime.h>

typedef __attribute__((ext_vector_type(16))) _Float16 v16h;
typedef __attribute__((ext_vector_type(8)))  _Float16 v8h;
typedef __attribute__((ext_vector_type(8)))  float    v8f;
typedef int v4i __attribute__((vector_size(16)));

#define N_ROWS 32768
#define K_CODES 8192
#define D_DIM 512
#define EPS 1e-5f

// ---- workspace layout (bytes) ----
#define ACC_OFF     0u           // 1024 f32 (sum[512], sumsq[512])
#define MEAN_OFF    4096u        // 512 f32
#define RSIG_OFF    6144u        // 512 f32
#define OSCALE_OFF  8192u        // 512 f32
#define HENORM_OFF  10240u       // 8192 f32
#define COUNTS_OFF  43008u       // 8192 f32
#define IDX_OFF     75776u       // 32768 i32
#define XN_OFF      206848u      // 32768*512 f16 = 32 MB
#define EB_OFF      33761280u    // 8192*512 f16 = 8 MB
#define DW_OFF      42149888u    // 8192*512 f32 = 16 MB

#if __has_builtin(__builtin_amdgcn_global_load_async_to_lds_b128)
#define HAVE_ASYNC_LDS 1
#else
#define HAVE_ASYNC_LDS 0
#endif

#if __has_builtin(__builtin_amdgcn_s_wait_asynccnt)
#define WAIT_ASYNC0() __builtin_amdgcn_s_wait_asynccnt(0)
#else
#define WAIT_ASYNC0() asm volatile("s_wait_asynccnt 0x0" ::: "memory")
#endif

__global__ void zero_f32(float* __restrict__ p, int n) {
  int i = blockIdx.x * blockDim.x + threadIdx.x;
  int stride = gridDim.x * blockDim.x;
  for (; i < n; i += stride) p[i] = 0.0f;
}

// per-block partial column sums over 128 rows, atomically merged
__global__ void stats_partial(const float* __restrict__ x, float* __restrict__ acc) {
  const int col = threadIdx.x;              // cols col and col+256
  const int rowbase = blockIdx.x * 128;
  float s0 = 0.f, q0 = 0.f, s1 = 0.f, q1 = 0.f;
  for (int r = 0; r < 128; ++r) {
    const float* row = x + (size_t)(rowbase + r) * D_DIM;
    float v0 = row[col], v1 = row[col + 256];
    s0 += v0; q0 += v0 * v0;
    s1 += v1; q1 += v1 * v1;
  }
  atomicAdd(&acc[col], s0);
  atomicAdd(&acc[col + 256], s1);
  atomicAdd(&acc[512 + col], q0);
  atomicAdd(&acc[512 + col + 256], q1);
}

__global__ void stats_finalize(const float* __restrict__ acc,
                               float* __restrict__ mean,
                               float* __restrict__ rsig,
                               float* __restrict__ oscale) {
  int d = blockIdx.x * 256 + threadIdx.x;
  float m = acc[d] * (1.0f / N_ROWS);
  float var = acc[512 + d] * (1.0f / N_ROWS) - m * m;
  mean[d] = m;
  rsig[d] = rsqrtf(var + EPS);
  // running var after one cumulative-average step = unbiased batch var
  oscale[d] = sqrtf(var * ((float)N_ROWS / (N_ROWS - 1.0f)) + EPS);
}

__global__ void normalize_x(const float* __restrict__ x,
                            const float* __restrict__ mean,
                            const float* __restrict__ rsig,
                            _Float16* __restrict__ xn) {
  size_t i = (size_t)blockIdx.x * 256 + threadIdx.x;  // covers N*D
  int d = (int)(i & (D_DIM - 1));
  xn[i] = (_Float16)((x[i] - mean[d]) * rsig[d]);
}

__global__ void prep_codebook(const float* __restrict__ e,
                              _Float16* __restrict__ eb,
                              float* __restrict__ henorm) {
  __shared__ float red[256];
  const int k = blockIdx.x;
  const float* row = e + (size_t)k * D_DIM;
  float v0 = row[threadIdx.x], v1 = row[threadIdx.x + 256];
  eb[(size_t)k * D_DIM + threadIdx.x] = (_Float16)v0;
  eb[(size_t)k * D_DIM + threadIdx.x + 256] = (_Float16)v1;
  red[threadIdx.x] = v0 * v0 + v1 * v1;
  __syncthreads();
  for (int s = 128; s > 0; s >>= 1) {
    if (threadIdx.x < s) red[threadIdx.x] += red[threadIdx.x + s];
    __syncthreads();
  }
  if (threadIdx.x == 0) henorm[k] = 0.5f * red[0];
}

// ---------------------------------------------------------------------------
// WMMA nearest-code search: argmax over k of (x . e_k - 0.5*||e_k||^2)
// block = 256 threads = 8 waves; each wave owns 16 rows of X, register-resident
// (X read from HBM exactly once). The 16-code B tile (16 KB) is staged into
// LDS ONCE per block via async global->LDS (double-buffered, ASYNCcnt), then
// all 8 waves read fragments with ds_load_b128 from a padded layout:
// row stride 520 halfs (1040 B) -> dword bank = 4*row + ... -> each 16-lane
// phase of a b128 read hits 16 disjoint 4-bank groups (conflict-free).
// ---------------------------------------------------------------------------
#define TILE_HALFS_PAD 520             // 512 data halfs + 8 pad halfs per row
#define TILE_SZ        (16 * TILE_HALFS_PAD)

__global__ void __launch_bounds__(256)
argmin_wmma(const _Float16* __restrict__ xn,
            const _Float16* __restrict__ eb,
            const float* __restrict__ henorm,
            int* __restrict__ idxout) {
#if HAVE_ASYNC_LDS
  __shared__ _Float16 bsm[2 * TILE_SZ];  // 33,280 B double buffer
#endif
  const int tid  = threadIdx.x;
  const int lane = tid & 31;
  const int wave = tid >> 5;
  const int half = lane >> 4;    // 0 or 1
  const int l16  = lane & 15;
  const int rowbase = blockIdx.x * 128 + wave * 16;

  // ---- A fragments: 16 chunks of K=32 over D=512, contiguous k order (k = 16*half + s) ----
  // lane (half, l16) holds row M=l16; chunk c: half0 -> K [0..15],
  // half1 -> K [16..31] => one 32-byte load per chunk.
  v16h a[16];
  {
    const _Float16* xrow = xn + (size_t)(rowbase + l16) * D_DIM;
#pragma unroll
    for (int c = 0; c < 16; ++c) {
      a[c] = *(const v16h*)(xrow + c * 32 + half * 16);
    }
  }

  float best[8];
  int bidx[8];
#pragma unroll
  for (int r = 0; r < 8; ++r) { best[r] = -3.0e38f; bidx[r] = 0; }

#if HAVE_ASYNC_LDS
  typedef __attribute__((address_space(1))) v4i gm_v4i;
  typedef __attribute__((address_space(3))) v4i lds_v4i;
  // stage tile t (16 codes x 1024 B) into buffer `buf`: 1024 16-B chunks,
  // 256 threads x 4 async transfers each
  auto issue_stage = [&](int t, int buf) {
    const _Float16* gbase = eb + (size_t)t * 16 * D_DIM;
#pragma unroll
    for (int i = 0; i < 4; ++i) {
      const int j   = tid + i * 256;       // chunk id 0..1023
      const int row = j >> 6;              // 64 chunks per code row
      const int col = (j & 63) * 8;        // half offset within row
      __builtin_amdgcn_global_load_async_to_lds_b128(
          (gm_v4i*)(gbase + row * D_DIM + col),
          (lds_v4i*)&bsm[buf * TILE_SZ + row * TILE_HALFS_PAD + col],
          0, 0);
    }
  };

  issue_stage(0, 0);
  for (int t = 0; t < K_CODES / 16; ++t) {
    WAIT_ASYNC0();        // this wave's stage-t transfers have landed
    __syncthreads();      // all waves' transfers landed; prev buffer free
    if (t + 1 < K_CODES / 16) issue_stage(t + 1, (t + 1) & 1);

    const int code  = t * 16 + l16;        // B column N=l16 == eb row `code`
    const int bbase = (t & 1) * TILE_SZ + l16 * TILE_HALFS_PAD;
    v8f acc = {};
#pragma unroll
    for (int ch = 0; ch < 16; ++ch) {
      // contiguous k order (same as A): lane holds column N=l16, half0 -> K[0..15],
      // half1 -> K[16..31] => two 16-byte ds loads.
      const _Float16* lp = &bsm[bbase + ch * 32 + half * 16];
      v8h lo = *(const v8h*)lp;
      v8h hi = *(const v8h*)(lp + 8);
      v16h bv = __builtin_shufflevector(lo, hi, 0, 1, 2, 3, 4, 5, 6, 7,
                                        8, 9, 10, 11, 12, 13, 14, 15);
      acc = __builtin_amdgcn_wmma_f32_16x16x32_f16(
          false, a[ch], false, bv, (short)0, acc, false, false);
    }

    const float hn = henorm[code];         // C-layout lane column is also l16
#pragma unroll
    for (int r = 0; r < 8; ++r) {
      float s = acc[r] - hn;
      if (s > best[r]) { best[r] = s; bidx[r] = code; }
    }
  }
#else
  // fallback: direct global loads of B fragments (all-L0-hit after wave 0)
  for (int t = 0; t < K_CODES / 16; ++t) {
    const int code = t * 16 + l16;
    const _Float16* erow = eb + (size_t)code * D_DIM;
    if (t + 1 < K_CODES / 16)
      __builtin_prefetch((const void*)(erow + 16 * D_DIM), 0, 1);
    v8f acc = {};
#pragma unroll
    for (int ch = 0; ch < 16; ++ch) {
      v16h bv = *(const v16h*)(erow + ch * 32 + half * 16);
      acc = __builtin_amdgcn_wmma_f32_16x16x32_f16(
          false, a[ch], false, bv, (short)0, acc, false, false);
    }
    const float hn = henorm[code];
#pragma unroll
    for (int r = 0; r < 8; ++r) {
      float s = acc[r] - hn;
      if (s > best[r]) { best[r] = s; bidx[r] = code; }
    }
  }
#endif

  // C layout: VGPR r -> row M = r + 8*half; lanes of one half cover all 16 cols
#pragma unroll
  for (int r = 0; r < 8; ++r) {
    float bv = best[r];
    int bi = bidx[r];
#pragma unroll
    for (int off = 8; off >= 1; off >>= 1) {
      float ov = __shfl_xor(bv, off, 16);
      int oi = __shfl_xor(bi, off, 16);
      if (ov > bv || (ov == bv && oi < bi)) { bv = ov; bi = oi; }
    }
    if (l16 == 0) idxout[rowbase + r + 8 * half] = bi;
  }
}

// dw[k,:] += xn[n,:], counts[k] += 1 via atomics (xn recomputed in fp32)
__global__ void scatter_dw(const float* __restrict__ x,
                           const float* __restrict__ mean,
                           const float* __restrict__ rsig,
                           const int* __restrict__ idx,
                           float* __restrict__ dw,
                           float* __restrict__ counts) {
  const int n = blockIdx.x;
  const int k = idx[n];
  const float* row = x + (size_t)n * D_DIM;
  float* drow = dw + (size_t)k * D_DIM;
  const int d0 = threadIdx.x, d1 = threadIdx.x + 256;
  atomicAdd(&drow[d0], (row[d0] - mean[d0]) * rsig[d0]);
  atomicAdd(&drow[d1], (row[d1] - mean[d1]) * rsig[d1]);
  if (threadIdx.x == 0) atomicAdd(&counts[k], 1.0f);
}

__global__ void finalize_out(const float* __restrict__ e,
                             const float* __restrict__ cs,
                             const float* __restrict__ dw,
                             const float* __restrict__ counts,
                             const float* __restrict__ mean,
                             const float* __restrict__ oscale,
                             float* __restrict__ out) {
  const int k = blockIdx.x;
  const float c_old = cs[k];
  const float inv = 1.0f / (c_old * 0.9f + 0.1f * counts[k]);
  const size_t base = (size_t)k * D_DIM;
  const int d0 = threadIdx.x, d1 = threadIdx.x + 256;
  out[base + d0] =
      (c_old * e[base + d0] * 0.9f + 0.1f * dw[base + d0]) * inv * oscale[d0] + mean[d0];
  out[base + d1] =
      (c_old * e[base + d1] * 0.9f + 0.1f * dw[base + d1]) * inv * oscale[d1] + mean[d1];
}

extern "C" void kernel_launch(void* const* d_in, const int* in_sizes, int n_in,
                              void* d_out, int out_size, void* d_ws, size_t ws_size,
                              hipStream_t stream) {
  const float* x  = (const float*)d_in[0];   // [N, D]
  const float* e  = (const float*)d_in[1];   // [K, D]
  const float* cs = (const float*)d_in[2];   // [K]
  float* out = (float*)d_out;                // [K, D]

  char* ws = (char*)d_ws;
  float*    acc    = (float*)(ws + ACC_OFF);
  float*    mean   = (float*)(ws + MEAN_OFF);
  float*    rsig   = (float*)(ws + RSIG_OFF);
  float*    oscale = (float*)(ws + OSCALE_OFF);
  float*    henorm = (float*)(ws + HENORM_OFF);
  float*    counts = (float*)(ws + COUNTS_OFF);
  int*      idxb   = (int*)(ws + IDX_OFF);
  _Float16* xnh    = (_Float16*)(ws + XN_OFF);
  _Float16* ebh    = (_Float16*)(ws + EB_OFF);
  float*    dw     = (float*)(ws + DW_OFF);

  zero_f32<<<8, 256, 0, stream>>>(acc, 1024);
  zero_f32<<<32, 256, 0, stream>>>(counts, K_CODES);
  zero_f32<<<16384, 256, 0, stream>>>(dw, K_CODES * D_DIM);

  stats_partial<<<N_ROWS / 128, 256, 0, stream>>>(x, acc);
  stats_finalize<<<2, 256, 0, stream>>>(acc, mean, rsig, oscale);

  normalize_x<<<(N_ROWS * D_DIM) / 256, 256, 0, stream>>>(x, mean, rsig, xnh);
  prep_codebook<<<K_CODES, 256, 0, stream>>>(e, ebh, henorm);

  argmin_wmma<<<N_ROWS / 128, 256, 0, stream>>>(xnh, ebh, henorm, idxb);

  scatter_dw<<<N_ROWS, 256, 0, stream>>>(x, mean, rsig, idxb, dw, counts);
  finalize_out<<<K_CODES, 256, 0, stream>>>(e, cs, dw, counts, mean, oscale, out);
}
